// Attention_43963285242705
// MI455X (gfx1250) — matrix operand order repaired and checked by reading, not compiled, not run on hardware
//
#include <hip/hip_runtime.h>
#include <hip/hip_bf16.h>

// ---------------- problem constants ----------------
#define BB   2
#define LL   2048
#define DD   4096
#define HH   32
#define KVH  8
#define HD   128
#define TOK  (BB * LL)           // 4096
#define SCALE 0.08838834764831845f   // 128^-0.5
#define THETA 10000.0f
#define NEG_INF (-1.0e30f)

// ---------------- WMMA fragment types ----------------
typedef __attribute__((ext_vector_type(16))) __bf16 bf16x16;
typedef __attribute__((ext_vector_type(8)))  __bf16 bf16x8;
typedef __attribute__((ext_vector_type(8)))  float  f32x8;

union BF16Frag {
    bf16x16 v;
    bf16x8  h[2];
};

__device__ __forceinline__ f32x8 wmma_bf16(const BF16Frag& a, const BF16Frag& b, f32x8 c) {
    return __builtin_amdgcn_wmma_f32_16x16x32_bf16(false, a.v, false, b.v,
                                                   (short)0, c, false, false);
}

__device__ __forceinline__ float redmax16(float v) {
    #pragma unroll
    for (int m = 1; m < 16; m <<= 1) v = fmaxf(v, __shfl_xor(v, m, 32));
    return v;
}
__device__ __forceinline__ float redsum16(float v) {
    #pragma unroll
    for (int m = 1; m < 16; m <<= 1) v += __shfl_xor(v, m, 32);
    return v;
}

// ---- CDNA5 async global->LDS copy (ASYNCcnt path), 16B per lane ----
__device__ __forceinline__ void async_b128(const void* smem, const __bf16* gptr) {
    unsigned lds_off = (unsigned)(size_t)smem;   // low 32 bits of flat LDS ptr = LDS offset
    unsigned long long ga = (unsigned long long)(size_t)gptr;
    asm volatile("global_load_async_to_lds_b128 %0, %1, off"
                 :: "v"(lds_off), "v"(ga) : "memory");
}
__device__ __forceinline__ void wait_async0() {
    asm volatile("s_wait_asynccnt 0x0" ::: "memory");
}

// ============================================================
// fp32 -> bf16 flat convert
// ============================================================
__global__ void cvt_bf16(const float* __restrict__ in, __bf16* __restrict__ out, int n) {
    int i = blockIdx.x * blockDim.x + threadIdx.x;
    if (i < n) out[i] = (__bf16)in[i];
}

// ============================================================
// W[K][N] fp32  ->  Wt[N][K] bf16   (32x32 LDS tiles, 256 thr)
// ============================================================
__global__ __launch_bounds__(256)
void transpose_cvt_bf16(const float* __restrict__ W, __bf16* __restrict__ Wt,
                        int K, int N) {
    __shared__ float t[32][33];
    const int bx = blockIdx.x * 32;   // n base
    const int by = blockIdx.y * 32;   // k base
    const int tx = threadIdx.x & 31;
    const int ty = threadIdx.x >> 5;  // 0..7
    #pragma unroll
    for (int i = 0; i < 32; i += 8)
        t[ty + i][tx] = W[(size_t)(by + ty + i) * N + bx + tx];
    __syncthreads();
    #pragma unroll
    for (int i = 0; i < 32; i += 8)
        Wt[(size_t)(bx + ty + i) * K + by + tx] = (__bf16)t[tx][ty + i];
}

// ============================================================
// bf16 WMMA GEMM with double-buffered async global->LDS staging.
//   C[M,N] = A[M,K] * Bt[N,K]^T (+ bias[N])
// block tile 128x128, 256 threads = 8 waves (4x2), wave tile 32x64.
// ============================================================
__global__ __launch_bounds__(256)
void gemm_bf16_async(const __bf16* __restrict__ A,   // [M][K]
                     const __bf16* __restrict__ Bt,  // [N][K]
                     const float* __restrict__ bias,
                     float* __restrict__ C,
                     int M, int N, int Kdim) {
    __shared__ __bf16 sA[2][128][32];   // [buf][m][k]  8KB each
    __shared__ __bf16 sB[2][128][32];   // [buf][n][k]  8KB each

    const int tid  = threadIdx.x;
    const int w    = tid >> 5;
    const int lane = tid & 31;
    const int lr   = lane & 15;
    const int hi   = lane >> 4;
    const int wm   = w & 3;
    const int wn   = w >> 2;
    const int blockN = blockIdx.x * 128;
    const int blockM = blockIdx.y * 128;

    f32x8 acc[2][4];
    const f32x8 zf = {0.f,0.f,0.f,0.f,0.f,0.f,0.f,0.f};
    #pragma unroll
    for (int r = 0; r < 2; ++r)
        #pragma unroll
        for (int c = 0; c < 4; ++c) acc[r][c] = zf;

    // async segment map: thread t copies rows r0 and r0+64, 16B at col c0
    const int r0 = tid >> 2;            // 0..63
    const int c0 = (tid & 3) * 8;       // bf16 col: 0,8,16,24
    const __bf16* gA = A  + (size_t)(blockM + r0) * Kdim + c0;
    const __bf16* gB = Bt + (size_t)(blockN + r0) * Kdim + c0;
    const size_t rowskip = (size_t)64 * Kdim;

    const int nsteps = Kdim >> 5;

    // prologue: stage k-step 0 into buffer 0
    async_b128(&sA[0][r0][c0],      gA);
    async_b128(&sA[0][r0 + 64][c0], gA + rowskip);
    async_b128(&sB[0][r0][c0],      gB);
    async_b128(&sB[0][r0 + 64][c0], gB + rowskip);

    for (int s = 0; s < nsteps; ++s) {
        const int buf = s & 1;
        wait_async0();          // this wave's copies for `buf` landed
        __syncthreads();        // all waves landed; prev readers of buf^1 done

        if (s + 1 < nsteps) {   // overlap next tile's DMA with this tile's WMMAs
            const int k1 = (s + 1) << 5;
            async_b128(&sA[buf ^ 1][r0][c0],      gA + k1);
            async_b128(&sA[buf ^ 1][r0 + 64][c0], gA + rowskip + k1);
            async_b128(&sB[buf ^ 1][r0][c0],      gB + k1);
            async_b128(&sB[buf ^ 1][r0 + 64][c0], gB + rowskip + k1);
        }

        BF16Frag afr[2];
        #pragma unroll
        for (int r = 0; r < 2; ++r) {
            const __bf16* p = &sA[buf][wm * 32 + r * 16 + lr][hi * 8];
            afr[r].h[0] = *(const bf16x8*)p;
            afr[r].h[1] = *(const bf16x8*)(p + 16);
        }
        BF16Frag bfr[4];
        #pragma unroll
        for (int c = 0; c < 4; ++c) {
            const __bf16* p = &sB[buf][wn * 64 + c * 16 + lr][hi * 8];
            bfr[c].h[0] = *(const bf16x8*)p;
            bfr[c].h[1] = *(const bf16x8*)(p + 16);
        }
        #pragma unroll
        for (int r = 0; r < 2; ++r)
            #pragma unroll
            for (int c = 0; c < 4; ++c)
                acc[r][c] = wmma_bf16(afr[r], bfr[c], acc[r][c]);
    }

    #pragma unroll
    for (int r = 0; r < 2; ++r) {
        #pragma unroll
        for (int c = 0; c < 4; ++c) {
            const int col = blockN + wn * 64 + c * 16 + lr;
            const float badd = bias ? bias[col] : 0.f;
            #pragma unroll
            for (int i = 0; i < 8; ++i) {
                const int row = blockM + wm * 32 + r * 16 + hi * 8 + i;
                C[(size_t)row * N + col] = acc[r][c][i] + badd;
            }
        }
    }
}

// ============================================================
// Half-split RoPE, in place on [TOK, nheads*HD] fp32 buffer.
// ============================================================
__global__ void rope_kernel(float* __restrict__ buf, int nheads, int total) {
    int idx = blockIdx.x * blockDim.x + threadIdx.x;
    if (idx >= total) return;
    const int i = idx & 63;
    const int h = (idx >> 6) % nheads;
    const int t = idx / (64 * nheads);
    const int pos = t % LL;
    const float inv = __powf(THETA, -(float)(2 * i) / (float)HD);
    float s, c;
    __sincosf((float)pos * inv, &s, &c);
    float* p = buf + (size_t)t * (nheads * HD) + h * HD + i;
    const float x1 = p[0], x2 = p[64];
    p[0]  = x1 * c - x2 * s;
    p[64] = x2 * c + x1 * s;
}

// ============================================================
// Flash-style GQA attention; fp32 Q/K/V in, bf16 output.
// grid = (B*H, L/128), 256 threads = 8 waves; wave owns 16 queries.
// ============================================================
__global__ __launch_bounds__(256)
void attn_wmma(const float* __restrict__ Q, const float* __restrict__ K,
               const float* __restrict__ V, __bf16* __restrict__ O) {
    __shared__ __bf16 sK [32][HD];     // [key][dim]
    __shared__ __bf16 sVT[HD][32];     // [dim][key]
    __shared__ __bf16 sP [8][16][32];  // per-wave probs [q][key]

    const int bh = blockIdx.x;
    const int b  = bh >> 5;
    const int h  = bh & 31;
    const int kvh = h >> 2;
    const int qbase = blockIdx.y * 128;

    const int tid  = threadIdx.x;
    const int w    = tid >> 5;
    const int lane = tid & 31;
    const int lr   = lane & 15;
    const int hi   = lane >> 4;

    BF16Frag aq[4];
    {
        const float* qp = Q + (size_t)(b * LL + qbase + w * 16 + lr) * (HH * HD) + h * HD;
        #pragma unroll
        for (int kk = 0; kk < 4; ++kk) {
            const int kb = kk * 32 + hi * 8;
            #pragma unroll
            for (int e = 0; e < 8; ++e) {
                aq[kk].h[0][e] = (__bf16)(qp[kb + e]      * SCALE);
                aq[kk].h[1][e] = (__bf16)(qp[kb + 16 + e] * SCALE);
            }
        }
    }

    const f32x8 zf = {0.f,0.f,0.f,0.f,0.f,0.f,0.f,0.f};
    f32x8 acc[8];
    #pragma unroll
    for (int j = 0; j < 8; ++j) acc[j] = zf;
    float mrow[8], lrow[8];
    #pragma unroll
    for (int i = 0; i < 8; ++i) { mrow[i] = NEG_INF; lrow[i] = 0.f; }

    const float* Kbase = K + (size_t)b * LL * (KVH * HD) + kvh * HD;
    const float* Vbase = V + (size_t)b * LL * (KVH * HD) + kvh * HD;
    const int ldk = tid >> 3;
    const int ldd = (tid & 7) * 16;

    for (int c0 = 0; c0 < LL; c0 += 32) {
        __syncthreads();
        {
            const float* kp = Kbase + (size_t)(c0 + ldk) * (KVH * HD) + ldd;
            const float* vp = Vbase + (size_t)(c0 + ldk) * (KVH * HD) + ldd;
            #pragma unroll
            for (int j = 0; j < 16; ++j) {
                sK[ldk][ldd + j]  = (__bf16)kp[j];
                sVT[ldd + j][ldk] = (__bf16)vp[j];
            }
        }
        __syncthreads();

        f32x8 s0 = zf, s1 = zf;
        #pragma unroll
        for (int kk = 0; kk < 4; ++kk) {
            BF16Frag b0, b1;
            const __bf16* p0 = &sK[lr][kk * 32 + hi * 8];
            b0.h[0] = *(const bf16x8*)p0;  b0.h[1] = *(const bf16x8*)(p0 + 16);
            const __bf16* p1 = &sK[16 + lr][kk * 32 + hi * 8];
            b1.h[0] = *(const bf16x8*)p1;  b1.h[1] = *(const bf16x8*)(p1 + 16);
            s0 = wmma_bf16(aq[kk], b0, s0);
            s1 = wmma_bf16(aq[kk], b1, s1);
        }

        #pragma unroll
        for (int i = 0; i < 8; ++i) {
            float cm = redmax16(fmaxf(s0[i], s1[i]));
            const float mnew = fmaxf(mrow[i], cm);
            const float fs = __expf(mrow[i] - mnew);
            mrow[i] = mnew;
            const float p0 = __expf(s0[i] - mnew);
            const float p1 = __expf(s1[i] - mnew);
            lrow[i] = lrow[i] * fs + redsum16(p0 + p1);
            #pragma unroll
            for (int j = 0; j < 8; ++j) acc[j][i] *= fs;
            sP[w][hi * 8 + i][lr]      = (__bf16)p0;
            sP[w][hi * 8 + i][16 + lr] = (__bf16)p1;
        }
        __syncthreads();

        BF16Frag ap;
        {
            const __bf16* pp = &sP[w][lr][hi * 8];
            ap.h[0] = *(const bf16x8*)pp;
            ap.h[1] = *(const bf16x8*)(pp + 16);
        }
        #pragma unroll
        for (int j = 0; j < 8; ++j) {
            BF16Frag bv;
            const __bf16* vp = &sVT[j * 16 + lr][hi * 8];
            bv.h[0] = *(const bf16x8*)vp;  bv.h[1] = *(const bf16x8*)(vp + 16);
            acc[j] = wmma_bf16(ap, bv, acc[j]);
        }
    }

    #pragma unroll
    for (int i = 0; i < 8; ++i) {
        const int qrow = qbase + w * 16 + hi * 8 + i;
        const float inv = 1.f / lrow[i];
        __bf16* orow = O + (size_t)(b * LL + qrow) * (HH * HD) + h * HD;
        #pragma unroll
        for (int j = 0; j < 8; ++j) orow[j * 16 + lr] = (__bf16)(acc[j][i] * inv);
    }
}

// ============================================================
extern "C" void kernel_launch(void* const* d_in, const int* in_sizes, int n_in,
                              void* d_out, int out_size, void* d_ws, size_t ws_size,
                              hipStream_t stream) {
    const float* x  = (const float*)d_in[0];
    const float* Wq = (const float*)d_in[1];
    const float* bq = (const float*)d_in[2];
    const float* Wk = (const float*)d_in[3];
    const float* bk = (const float*)d_in[4];
    const float* Wv = (const float*)d_in[5];
    const float* bv = (const float*)d_in[6];
    const float* Wo = (const float*)d_in[7];
    float* out = (float*)d_out;

    // ---- workspace layout ----
    char* ws = (char*)d_ws;
    __bf16* xbf  = (__bf16*)ws;                 ws += (size_t)TOK * DD * 2;         // 33.5MB
    __bf16* WqT  = (__bf16*)ws;                 ws += (size_t)DD * (HH * HD) * 2;   // 33.5MB
    __bf16* WkT  = (__bf16*)ws;                 ws += (size_t)DD * (KVH * HD) * 2;  // 8.4MB
    __bf16* WvT  = (__bf16*)ws;                 ws += (size_t)DD * (KVH * HD) * 2;  // 8.4MB
    __bf16* WoT  = (__bf16*)ws;                 ws += (size_t)DD * DD * 2;          // 33.5MB
    float*  qbuf = (float*)ws;                  ws += (size_t)TOK * (HH * HD) * 4;  // 64MB
    float*  kbuf = (float*)ws;                  ws += (size_t)TOK * (KVH * HD) * 4; // 16MB
    float*  vbuf = (float*)ws;                  ws += (size_t)TOK * (KVH * HD) * 4; // 16MB
    __bf16* obf  = (__bf16*)ws;                                                     // 33.5MB

    // ---- one-shot precision/layout prep (memory-bound, ~15us) ----
    {
        int n = TOK * DD;
        cvt_bf16<<<(n + 255) / 256, 256, 0, stream>>>(x, xbf, n);
        transpose_cvt_bf16<<<dim3((HH * HD) / 32, DD / 32), 256, 0, stream>>>(Wq, WqT, DD, HH * HD);
        transpose_cvt_bf16<<<dim3((KVH * HD) / 32, DD / 32), 256, 0, stream>>>(Wk, WkT, DD, KVH * HD);
        transpose_cvt_bf16<<<dim3((KVH * HD) / 32, DD / 32), 256, 0, stream>>>(Wv, WvT, DD, KVH * HD);
        transpose_cvt_bf16<<<dim3(DD / 32, DD / 32), 256, 0, stream>>>(Wo, WoT, DD, DD);
    }

    // ---- QKV projections: async-staged bf16 WMMA GEMMs ----
    gemm_bf16_async<<<dim3((HH * HD) / 128, TOK / 128), 256, 0, stream>>>(
        xbf, WqT, bq, qbuf, TOK, HH * HD, DD);
    gemm_bf16_async<<<dim3((KVH * HD) / 128, TOK / 128), 256, 0, stream>>>(
        xbf, WkT, bk, kbuf, TOK, KVH * HD, DD);
    gemm_bf16_async<<<dim3((KVH * HD) / 128, TOK / 128), 256, 0, stream>>>(
        xbf, WvT, bv, vbuf, TOK, KVH * HD, DD);

    // ---- RoPE in place on Q and K (fp32 precision) ----
    {
        int totq = TOK * HH * 64;
        rope_kernel<<<(totq + 255) / 256, 256, 0, stream>>>(qbuf, HH, totq);
        int totk = TOK * KVH * 64;
        rope_kernel<<<(totk + 255) / 256, 256, 0, stream>>>(kbuf, KVH, totk);
    }

    // ---- GQA flash attention (writes bf16 for the O-projection) ----
    attn_wmma<<<dim3(BB * HH, LL / 128), 256, 0, stream>>>(qbuf, kbuf, vbuf, obf);

    // ---- Output projection ----
    gemm_bf16_async<<<dim3(DD / 128, TOK / 128), 256, 0, stream>>>(
        obf, WoT, nullptr, out, TOK, DD, DD);
}
